// RGCN_28853590295095
// MI455X (gfx1250) — matrix operand order repaired and checked by reading, not compiled, not run on hardware
//
#include <hip/hip_runtime.h>
#include <hip/hip_bf16.h>

typedef __bf16 bf16;
typedef __attribute__((ext_vector_type(4)))  __bf16 v4bf;
typedef __attribute__((ext_vector_type(8)))  __bf16 v8bf;
typedef __attribute__((ext_vector_type(16))) __bf16 v16bf;
typedef __attribute__((ext_vector_type(8)))  float  v8f;
typedef __attribute__((ext_vector_type(4)))  int    v4i;

#define AS1 __attribute__((address_space(1)))
#define AS3 __attribute__((address_space(3)))

#define RELS  8
#define DIM   256
#define PANEL 8192              // bf16 elements per 32x256 k-panel (16 KB)

// ---- async global->LDS 16-byte copy (CDNA5 GLOBAL_LOAD_ASYNC_TO_LDS_B128)
__device__ __forceinline__ void async_copy16(const bf16* g, bf16* l) {
#if __has_builtin(__builtin_amdgcn_global_load_async_to_lds_b128)
  __builtin_amdgcn_global_load_async_to_lds_b128(
      (AS1 v4i*)(v4i*)(void*)const_cast<bf16*>(g),
      (AS3 v4i*)(v4i*)(void*)l,
      0, 0);
#else
  unsigned ldsOff = (unsigned)(size_t)(void*)l;
  asm volatile("global_load_async_to_lds_b128 %0, %1, off"
               :: "v"(ldsOff), "v"((size_t)(const void*)g) : "memory");
#endif
}

__device__ __forceinline__ void wait_async0() {
#if __has_builtin(__builtin_amdgcn_s_wait_asynccnt)
  __builtin_amdgcn_s_wait_asynccnt(0);
#else
  asm volatile("s_wait_asynccnt 0" ::: "memory");
#endif
}

// ---------------- W[r] = sum_b coef[r,b] * V[b]
// Stored bf16 in WMMA-B-swizzled panel layout:
//   W'[r][ks][ntile][h][lane][c]  (ks = i>>5, k = i&31, ntile = o>>4, h = k>>4, lane = ((k>>3)&1)*16 + (o&15), c = k&7)
// so a k-panel is one contiguous 16 KB chunk and every B-fragment LDS read is a
// per-lane-contiguous, bank-conflict-free 16 B load.
__global__ void __launch_bounds__(256)
k_combine_w(bf16* __restrict__ Wout, const float* __restrict__ coef,
            const float* __restrict__ V) {
  int idx = blockIdx.x * blockDim.x + threadIdx.x;   // r*65536 + i*256 + o
  int o   = idx & (DIM - 1);
  int i   = (idx >> 8) & (DIM - 1);
  int r   = idx >> 16;
  float s = 0.f;
#pragma unroll
  for (int b = 0; b < RELS; ++b)
    s += coef[r * RELS + b] * V[(size_t)b * DIM * DIM + (i << 8) + o];
  int ks = i >> 5, k = i & 31;
  int ntile = o >> 4, h = k >> 4, c = k & 7;
  int ln = ((k >> 3) & 1) * 16 + (o & 15);
  size_t woff = ((size_t)r << 16) + (size_t)ks * PANEL
              + (size_t)(((ntile * 2 + h) * 32 + ln) * 8 + c);
  Wout[woff] = (bf16)s;
}

// ---------------- f32 -> bf16
__global__ void __launch_bounds__(256)
k_f32_to_bf16(bf16* __restrict__ dstp, const float* __restrict__ srcp, int n) {
  int i = blockIdx.x * blockDim.x + threadIdx.x;
  if (i < n) dstp[i] = (bf16)srcp[i];
}

__global__ void __launch_bounds__(256)
k_relu_to_bf16(bf16* __restrict__ dstp, const float* __restrict__ srcp, int n) {
  int i = blockIdx.x * blockDim.x + threadIdx.x;
  if (i < n) {
    float v = srcp[i];
    dstp[i] = (bf16)(v > 0.f ? v : 0.f);
  }
}

// ---------------- agg[n][o] = bias[o]
__global__ void __launch_bounds__(256)
k_bias_init(float* __restrict__ agg, const float* __restrict__ bias, int total) {
  int i = blockIdx.x * blockDim.x + threadIdx.x;
  if (i < total) agg[i] = bias[i & (DIM - 1)];
}

// ---------------- T[r] = A @ W[r]   (bf16 in, f32 accum via WMMA, bf16 out)
// Block = 256 threads (8 waves) handles 2 M-tiles x full N=256 for one relation.
// W[r] k-panels (32x256, pre-swizzled) are async-staged into a 2x16KB LDS double
// buffer shared by all 8 waves; panel ks+1 streams in while WMMAs consume ks.
__global__ void __launch_bounds__(256)
k_gemm_bf16(bf16* __restrict__ T, const bf16* __restrict__ A,
            const bf16* __restrict__ W, int nnodes) {
  __shared__ bf16 smem[2 * PANEL];                   // 32 KB

  const int mtiles = nnodes >> 4;
  const int mpairs = mtiles >> 1;
  const int r  = blockIdx.x / mpairs;
  const int mp = blockIdx.x % mpairs;

  const int tid    = threadIdx.x;
  const int wv     = tid >> 5;
  const int lane   = tid & 31;
  const int ml     = wv >> 2;                        // m-tile within pair
  const int nstrip = wv & 3;                         // 64-col strip

  const int mrow = lane & 15;
  const int half = lane >> 4;
  const bf16* __restrict__ Arow = A + (size_t)((mp * 2 + ml) * 16 + mrow) * DIM;
  const bf16* __restrict__ Wr   = W + ((size_t)r << 16);

  v8f acc[4] = {};

  // stage one 16 KB panel: 256 threads x 4 async b128 copies
  auto stage = [&](int ks, int b) {
    const bf16* g = Wr + (size_t)ks * PANEL + tid * 8;
    bf16*       l = &smem[b * PANEL + tid * 8];
#pragma unroll
    for (int it = 0; it < 4; ++it)
      async_copy16(g + it * 2048, l + it * 2048);
  };

  auto compute = [&](int ks, int b) {
    const int kb = ks * 32;
    union { v16bf v; v8bf h[2]; } af;
    af.h[0] = *(const v8bf*)(Arow + kb + 8 * half);
    af.h[1] = *(const v8bf*)(Arow + kb + 16 + 8 * half);
    const bf16* base = &smem[b * PANEL];
#pragma unroll
    for (int j = 0; j < 4; ++j) {
      const int ntile = nstrip * 4 + j;
      union { v16bf v; v8bf h[2]; } bfr;
      bfr.h[0] = *(const v8bf*)(base + ((ntile * 2 + 0) * 32 + lane) * 8);
      bfr.h[1] = *(const v8bf*)(base + ((ntile * 2 + 1) * 32 + lane) * 8);
      acc[j] = __builtin_amdgcn_wmma_f32_16x16x32_bf16(
          false, af.v, false, bfr.v, (short)0, acc[j], false, false);
    }
  };

  stage(0, 0);
  wait_async0();
  __syncthreads();

#pragma unroll
  for (int ks = 0; ks < 8; ++ks) {
    if (ks < 7) stage(ks + 1, (ks + 1) & 1);
    compute(ks, ks & 1);
    if (ks < 7) {
      wait_async0();
      __syncthreads();
    }
  }

  // C/D layout: VGPR v -> row = v + 8*half ; col = lane&15
  const int n0  = nstrip * 64;
  const int col = n0 + (lane & 15);
  size_t base = ((size_t)r * nnodes + (size_t)((mp * 2 + ml) * 16 + 8 * half)) * DIM + col;
#pragma unroll
  for (int v = 0; v < 8; ++v) {
    T[base + (size_t)v * DIM +  0] = (bf16)acc[0][v];
    T[base + (size_t)v * DIM + 16] = (bf16)acc[1][v];
    T[base + (size_t)v * DIM + 32] = (bf16)acc[2][v];
    T[base + (size_t)v * DIM + 48] = (bf16)acc[3][v];
  }
}

// ---------------- edge scatter: agg[dst] += T[etype][src]  (64 thr/edge, 4 cols/thr)
__global__ void __launch_bounds__(256)
k_scatter(float* __restrict__ agg, const bf16* __restrict__ T,
          const int* __restrict__ srcI, const int* __restrict__ dstI,
          const int* __restrict__ et, int nedges, int nnodes) {
  int t = blockIdx.x * blockDim.x + threadIdx.x;
  int e = t >> 6;
  if (e >= nedges) return;
  int c = (t & 63) * 4;
  size_t row = ((size_t)et[e] * nnodes + (size_t)srcI[e]) * DIM + c;
  v4bf m = *(const v4bf*)(T + row);
  float* out = agg + (size_t)dstI[e] * DIM + c;
  atomicAdd(out + 0, (float)m[0]);
  atomicAdd(out + 1, (float)m[1]);
  atomicAdd(out + 2, (float)m[2]);
  atomicAdd(out + 3, (float)m[3]);
}

// ---------------- column sums over nodes
__global__ void __launch_bounds__(256)
k_zero(float* __restrict__ p, int n) {
  int i = blockIdx.x * blockDim.x + threadIdx.x;
  if (i < n) p[i] = 0.f;
}

__global__ void __launch_bounds__(256)
k_colsum(float* __restrict__ colsum, const float* __restrict__ agg,
         int nnodes, int chunk) {
  int t  = threadIdx.x;
  int n0 = blockIdx.x * chunk;
  int n1 = n0 + chunk; if (n1 > nnodes) n1 = nnodes;
  float s = 0.f;
  for (int n = n0; n < n1; ++n) s += agg[(size_t)n * DIM + t];
  atomicAdd(&colsum[t], s);
}

// ---------------- out[c] = fc_b[c] + (colsum/N) . fc_w[c]
__global__ void __launch_bounds__(256)
k_final(float* __restrict__ outp, const float* __restrict__ colsum,
        const float* __restrict__ fcw, const float* __restrict__ fcb, int nnodes) {
  __shared__ float sh[256];
  int t = threadIdx.x;
  float hv = colsum[t] / (float)nnodes;
#pragma unroll
  for (int c = 0; c < 2; ++c) {
    sh[t] = hv * fcw[c * DIM + t];
    __syncthreads();
    for (int s = 128; s > 0; s >>= 1) {
      if (t < s) sh[t] += sh[t + s];
      __syncthreads();
    }
    if (t == 0) outp[c] = sh[0] + fcb[c];
    __syncthreads();
  }
}

extern "C" void kernel_launch(void* const* d_in, const int* in_sizes, int n_in,
                              void* d_out, int out_size, void* d_ws, size_t ws_size,
                              hipStream_t stream) {
  const float* features = (const float*)d_in[0];
  const int*   srcI     = (const int*)  d_in[1];
  const int*   dstI     = (const int*)  d_in[2];
  const int*   et       = (const int*)  d_in[3];
  const float* V1       = (const float*)d_in[4];
  const float* coef1    = (const float*)d_in[5];
  const float* b1       = (const float*)d_in[6];
  const float* V2       = (const float*)d_in[7];
  const float* coef2    = (const float*)d_in[8];
  const float* b2       = (const float*)d_in[9];
  const float* fcw      = (const float*)d_in[10];
  const float* fcb      = (const float*)d_in[11];
  float* outp = (float*)d_out;

  const int nnodes = in_sizes[0] / DIM;   // 20000
  const int nedges = in_sizes[1];         // 320000
  const int nh     = nnodes * DIM;

  // carve workspace
  char* ws = (char*)d_ws;
  size_t off = 0;
  auto carve = [&](size_t bytes) -> char* {
    char* p = ws + off;
    off = (off + bytes + 255) & ~(size_t)255;
    return p;
  };
  bf16*  Wbf    = (bf16*) carve((size_t)RELS * DIM * DIM * sizeof(bf16));
  bf16*  hbf    = (bf16*) carve((size_t)nh * sizeof(bf16));
  bf16*  T      = (bf16*) carve((size_t)RELS * nnodes * DIM * sizeof(bf16));
  float* agg    = (float*)carve((size_t)nh * sizeof(float));
  float* colsum = (float*)carve((size_t)DIM * sizeof(float));
  (void)ws_size; (void)n_in; (void)out_size;

  const int mtiles = nnodes / 16;
  const int mpairs = mtiles / 2;
  const int gemmBlocks    = RELS * mpairs;
  const int cvtBlocks     = (nh + 255) / 256;
  const int scatterBlocks = (nedges * 64 + 255) / 256;

  // -------- layer 1
  k_combine_w   <<<RELS * DIM * DIM / 256, 256, 0, stream>>>(Wbf, coef1, V1);
  k_f32_to_bf16 <<<cvtBlocks, 256, 0, stream>>>(hbf, features, nh);
  k_gemm_bf16   <<<gemmBlocks, 256, 0, stream>>>(T, hbf, Wbf, nnodes);
  k_bias_init   <<<cvtBlocks, 256, 0, stream>>>(agg, b1, nh);
  k_scatter     <<<scatterBlocks, 256, 0, stream>>>(agg, T, srcI, dstI, et, nedges, nnodes);
  k_relu_to_bf16<<<cvtBlocks, 256, 0, stream>>>(hbf, agg, nh);

  // -------- layer 2
  k_combine_w   <<<RELS * DIM * DIM / 256, 256, 0, stream>>>(Wbf, coef2, V2);
  k_gemm_bf16   <<<gemmBlocks, 256, 0, stream>>>(T, hbf, Wbf, nnodes);
  k_bias_init   <<<cvtBlocks, 256, 0, stream>>>(agg, b2, nh);
  k_scatter     <<<scatterBlocks, 256, 0, stream>>>(agg, T, srcI, dstI, et, nedges, nnodes);

  // -------- readout
  k_zero  <<<1, 256, 0, stream>>>(colsum, DIM);
  k_colsum<<<80, 256, 0, stream>>>(colsum, agg, nnodes, (nnodes + 79) / 80);
  k_final <<<1, 256, 0, stream>>>(outp, colsum, fcw, fcb, nnodes);
}
